// Torch_LSTM_Classifer_26585847562880
// MI455X (gfx1250) — matrix operand order repaired and checked by reading, not compiled, not run on hardware
//
#include <hip/hip_runtime.h>

// ---------------------------------------------------------------------------
// 2-layer LSTM (B=32, T=2048, D=256, H=512) + sigmoid FC head for MI455X.
//
// Persistent-weight cluster of 64 workgroups; WG k owns hidden units
// [8k, 8k+8) => 32 gate columns (i,f,g,o). Its [W_ih;W_hh] slice is converted
// to bf16 WMMA B-fragments once and then held in VGPRs (v16bf breg[KT]) for
// the whole scan. Per timestep:
//   1) stage A = [x_t | h_{t-1}] into LDS A-fragment layout, using
//      global_load_async_to_lds_b128 (ASYNCcnt) for bf16 chunks and a
//      convert path for layer-0's fp32 input
//   2) g = A @ Wsliceᵀ via v_wmma_f32_16x16x32_bf16 (fp32 accum, B in regs)
//   3) fp32 gate math (cell state in registers), h slice -> double-buffered
//      global exchange array (bf16)
//   4) device-scope release/acquire barrier across the 64 WGs
// Layer 0 fuses the input projection (extra K columns) and emits the full
// hidden sequence (bf16) consumed as layer 1's input.
// ---------------------------------------------------------------------------

typedef __bf16 bf16_t;
typedef __attribute__((ext_vector_type(16))) __bf16 v16bf;
typedef __attribute__((ext_vector_type(8)))  __bf16 v8bf;
typedef __attribute__((ext_vector_type(8)))  float  v8f;

#define BB 32
#define TT 2048
#define HH 512
#define NWG 64        // workgroups in the persistent cluster
#define NTHREADS 128  // 4 wave32 per WG

struct GBar { unsigned cnt; unsigned phase; unsigned pad[62]; };

__device__ __forceinline__ float fsig(float x) { return 1.f / (1.f + __expf(-x)); }
__device__ __forceinline__ float ftanh(float x) {
  x = fminf(fmaxf(x, -15.f), 15.f);
  float e = __expf(2.f * x);
  return (e - 1.f) / (e + 1.f);
}

// Async 16-byte global->LDS copy (per-lane addresses; tracked on ASYNCcnt).
// Generic shared-aperture addresses keep the LDS byte offset in the low 32
// bits (flat->LDS mapping truncates), so (unsigned)(size_t)ptr is the VDST.
__device__ __forceinline__ void async_copy16(void* lds_dst, const void* gsrc) {
  unsigned dsoff = (unsigned)(size_t)lds_dst;
  unsigned long long ga = (unsigned long long)(size_t)gsrc;
  asm volatile("global_load_async_to_lds_b128 %0, %1, off"
               :: "v"(dsoff), "v"(ga) : "memory");
}
__device__ __forceinline__ void wait_async0() {
  asm volatile("s_wait_asynccnt 0x0" ::: "memory");
}

// Release/acquire barrier across all NWG workgroups (co-resident: 64 WGs on a
// >200-WGP part). Sense tracked by a monotonically rising phase.
__device__ __forceinline__ void grid_barrier(GBar* bar, unsigned expected) {
  __threadfence();       // every thread: make h stores device-visible
  __syncthreads();
  if (threadIdx.x == 0) {
    unsigned prev = atomicAdd(&bar->cnt, 1u);
    if (prev == NWG - 1) {
      bar->cnt = 0;
      __threadfence();
      atomicAdd(&bar->phase, 1u);
    } else {
      while (atomicAdd(&bar->phase, 0u) < expected) __builtin_amdgcn_s_sleep(2);
    }
    __threadfence();     // acquire
  }
  __syncthreads();
}

// ---------------------------------------------------------------------------
// DIN = input width (256 layer 0, 512 layer 1); IN_F32: fp32 input (layer 0).
// ---------------------------------------------------------------------------
template <int DIN, bool IN_F32>
__global__ __launch_bounds__(NTHREADS)
void lstm_scan(const void* __restrict__ xin_v,
               const float* __restrict__ Wih,   // (4H x DIN) row-major f32
               const float* __restrict__ Whh,   // (4H x H)   row-major f32
               const float* __restrict__ bih,
               const float* __restrict__ bhh,
               bf16_t* __restrict__ hx,         // [2][B][H] double buffer
               bf16_t* __restrict__ out_seq,    // [B][T][H] or null
               float*  __restrict__ out_last,   // [B][H]    or null
               GBar* __restrict__ bar)
{
  constexpr int K      = DIN + HH;      // concat K dimension
  constexpr int KT     = K / 32;        // k-tiles of 32
  constexpr int WBYTES = 2 * KT * 1024; // B fragments (prologue only)
  constexpr int ABYTES = 2 * KT * 1024; // A fragments: 2 m-tiles x KT x 1KB
  constexpr int GSTR   = 36;            // padded f32 gate-buffer stride
  constexpr int GBYTES = BB * GSTR * 4;
  constexpr int SBYTES = (WBYTES > ABYTES + GBYTES) ? WBYTES : (ABYTES + GBYTES);
  __shared__ char smem[SBYTES];         // A region aliases the dead W region

  const int tid   = threadIdx.x;
  const int lane  = tid & 31;           // wave32
  const int wid   = tid >> 5;
  const int mtile = wid & 1;            // batch tile (rows 0-15 / 16-31)
  const int ntile = wid >> 1;           // 16-column tile (2 per WG)
  const int u0    = blockIdx.x * 8;     // first hidden unit owned by this WG

  // ---- Stage weight slice into LDS as WMMA B fragments (transient). -------
  // B-fragment (16-bit, 32x16): lane = n + 16*hi holds K [kt*32+16*hi, +16)
  // of column n => 32 contiguous bytes per lane.
  for (int blk = wid; blk < 2 * KT; blk += NTHREADS / 32) {
    int nt = blk / KT, kt = blk % KT;
    int n = lane & 15, hi = lane >> 4;
    int c = nt * 16 + n;                           // WG-local column 0..31
    int gidx = (c >> 3) * HH + u0 + (c & 7);       // global gate row
    int k0 = kt * 32 + hi * 16;
    const float* src = (k0 < DIN) ? (Wih + (size_t)gidx * DIN + k0)
                                  : (Whh + (size_t)gidx * HH + (k0 - DIN));
    v8f f0 = *(const v8f*)src;
    v8f f1 = *(const v8f*)(src + 8);
    v8bf b0 = __builtin_convertvector(f0, v8bf);
    v8bf b1 = __builtin_convertvector(f1, v8bf);
    v16bf w = __builtin_shufflevector(b0, b1, 0,1,2,3,4,5,6,7,8,9,10,11,12,13,14,15);
    *(v16bf*)(smem + (blk << 10) + (lane << 5)) = w;
  }

  // Bias sums for this thread's two (batch, unit) pairs: constant over time.
  float bsum[2][4];
  float cstate[2] = {0.f, 0.f};
  #pragma unroll
  for (int pp = 0; pp < 2; ++pp) {
    int p = tid + pp * NTHREADS;
    int ul = p & 7;
    #pragma unroll
    for (int q = 0; q < 4; ++q) {
      int gidx = q * HH + u0 + ul;
      bsum[pp][q] = bih[gidx] + bhh[gidx];
    }
  }
  __syncthreads();

  // ---- Hoist this wave's B fragments into registers for the whole scan. ---
  v16bf breg[KT];
  #pragma unroll
  for (int kt = 0; kt < KT; ++kt)
    breg[kt] = *(const v16bf*)(smem + ((ntile * KT + kt) << 10) + (lane << 5));
  __syncthreads();   // W region is dead; A staging may now overwrite it

  float* gbuf = (float*)(smem + ABYTES);

  for (int t = 0; t < TT; ++t) {
    const bf16_t* hxr = hx + (size_t)((t + 1) & 1) * (BB * HH);
    bf16_t*       hxw = hx + (size_t)(t & 1) * (BB * HH);

    // ---- stage A = [x_t | h_{t-1}] into LDS A-fragment layout -------------
    // A-fragment (16-bit, 16x32): lane = (m&15)+16*hi holds two 16B runs:
    // run0 = K [kt*32+16*hi, +8), run1 = K [kt*32+16*hi+8, +8).
    constexpr int KPC = K / 8;  // 16-byte chunks per row
    #pragma unroll
    for (int i = 0; i < KT; ++i) {           // 32*KPC/128 == KT iterations
      int ci = i * NTHREADS + tid;
      int m  = ci / KPC;                     // batch row
      int k0 = (ci - m * KPC) * 8;
      int kt = k0 >> 5;
      int r  = (k0 >> 3) & 3;
      int hi = r >> 1, run = r & 1;
      char* dst = smem + (((kt << 1) | (m >> 4)) << 10)
                  + (((m & 15) | (hi << 4)) << 5) + (run << 4);
      if (k0 < DIN) {
        if (IN_F32) {  // layer 0: convert fp32 -> bf16 while staging
          const float* xp = (const float*)xin_v + ((size_t)m * TT + t) * DIN + k0;
          v8bf val = __builtin_convertvector(*(const v8f*)xp, v8bf);
          *(v8bf*)dst = val;
          if (t + 1 < TT && (k0 & 24) == 0) __builtin_prefetch(xp + DIN, 0, 1);
        } else {       // layer 1: bf16 input, async copy straight to LDS
          const bf16_t* xp = (const bf16_t*)xin_v + ((size_t)m * TT + t) * DIN + k0;
          async_copy16(dst, xp);
          if (t + 1 < TT && (k0 & 24) == 0) __builtin_prefetch(xp + DIN, 0, 1);
        }
      } else if (t == 0) {
        v8bf val;
        #pragma unroll
        for (int q = 0; q < 8; ++q) val[q] = (bf16_t)0.f;
        *(v8bf*)dst = val;
      } else {
        async_copy16(dst, hxr + m * HH + (k0 - DIN));
      }
    }
    wait_async0();
    __syncthreads();

    // ---- GEMM: one 16x16 tile per wave; A from LDS, B from registers ------
    v8f acc;
    #pragma unroll
    for (int q = 0; q < 8; ++q) acc[q] = 0.f;
    #pragma unroll
    for (int kt = 0; kt < KT; ++kt) {
      v16bf a = *(const v16bf*)(smem + (((kt << 1) | mtile) << 10) + (lane << 5));
      acc = __builtin_amdgcn_wmma_f32_16x16x32_bf16(
              false, a, false, breg[kt], (short)0, acc, false, false);
    }
    // scatter accumulator to the fp32 gate buffer (C layout: vgpr v -> row)
    {
      int colg = ntile * 16 + (lane & 15);
      int rb   = mtile * 16 + (lane >> 4) * 8;
      #pragma unroll
      for (int v = 0; v < 8; ++v) gbuf[(rb + v) * GSTR + colg] = acc[v];
    }
    __syncthreads();

    // ---- gate nonlinearity + state update (fp32) --------------------------
    #pragma unroll
    for (int pp = 0; pp < 2; ++pp) {
      int p  = tid + pp * NTHREADS;     // 0..255 -> (batch, unit-local)
      int b  = p >> 3, ul = p & 7;
      const float* gr = gbuf + b * GSTR;
      float gi = gr[ul]      + bsum[pp][0];
      float gf = gr[8 + ul]  + bsum[pp][1];
      float gc = gr[16 + ul] + bsum[pp][2];
      float go = gr[24 + ul] + bsum[pp][3];
      float i_ = fsig(gi), f_ = fsig(gf), c_ = ftanh(gc), o_ = fsig(go);
      float c = f_ * cstate[pp] + i_ * c_;
      cstate[pp] = c;
      float h = o_ * ftanh(c);
      int ug = u0 + ul;
      hxw[b * HH + ug] = (bf16_t)h;
      if (out_seq) out_seq[((size_t)b * TT + t) * HH + ug] = (bf16_t)h;
      if (out_last && t == TT - 1) out_last[b * HH + ug] = h;
    }

    grid_barrier(bar, (unsigned)(t + 1));
  }
}

// ---------------------------------------------------------------------------
__global__ void k_init(GBar* bars) {
  int i = threadIdx.x;
  if (i < 2) { bars[i].cnt = 0; bars[i].phase = 0; }
}

__global__ void k_fc(const float* __restrict__ hlast,
                     const float* __restrict__ Wfc,
                     const float* __restrict__ bfc,
                     float* __restrict__ out) {
  int b = threadIdx.x;
  if (b >= BB) return;
  float s = 0.f;
  #pragma unroll 8
  for (int j = 0; j < HH; ++j) s += hlast[b * HH + j] * Wfc[j];
  out[b] = fsig(s + bfc[0]);
}

// ---------------------------------------------------------------------------
extern "C" void kernel_launch(void* const* d_in, const int* in_sizes, int n_in,
                              void* d_out, int out_size, void* d_ws, size_t ws_size,
                              hipStream_t stream) {
  const float* x    = (const float*)d_in[0];
  const float* Wih0 = (const float*)d_in[1];
  const float* Whh0 = (const float*)d_in[2];
  const float* bih0 = (const float*)d_in[3];
  const float* bhh0 = (const float*)d_in[4];
  const float* Wih1 = (const float*)d_in[5];
  const float* Whh1 = (const float*)d_in[6];
  const float* bih1 = (const float*)d_in[7];
  const float* bhh1 = (const float*)d_in[8];
  const float* Wfc  = (const float*)d_in[9];
  const float* bfc  = (const float*)d_in[10];

  char* ws = (char*)d_ws;
  size_t off = 0;
  auto carve = [&](size_t bytes) -> char* {
    char* p = ws + off;
    off += (bytes + 255) & ~(size_t)255;
    return p;
  };
  bf16_t* h1seq  = (bf16_t*)carve((size_t)BB * TT * HH * sizeof(bf16_t)); // 64 MiB
  bf16_t* hx     = (bf16_t*)carve((size_t)2 * BB * HH * sizeof(bf16_t));
  float*  h2last = (float*) carve((size_t)BB * HH * sizeof(float));
  GBar*   bars   = (GBar*)  carve(2 * sizeof(GBar));

  k_init<<<1, 64, 0, stream>>>(bars);
  lstm_scan<256, true ><<<NWG, NTHREADS, 0, stream>>>(
      x, Wih0, Whh0, bih0, bhh0, hx, h1seq, nullptr, &bars[0]);
  lstm_scan<512, false><<<NWG, NTHREADS, 0, stream>>>(
      h1seq, Wih1, Whh1, bih1, bhh1, hx, nullptr, h2last, &bars[1]);
  k_fc<<<1, 32, 0, stream>>>(h2last, Wfc, bfc, (float*)d_out);
}
